// TransformerEncoderBlock_9801115370093
// MI455X (gfx1250) — compile-verified, matrix operand order checked by reading, not run on hardware
//
#include <hip/hip_runtime.h>
#include <math.h>

typedef __attribute__((ext_vector_type(16))) __bf16 v16bf;
typedef __attribute__((ext_vector_type(8)))  __bf16 bf16x8;
typedef __attribute__((ext_vector_type(4)))  __bf16 bf16x4;
typedef __attribute__((ext_vector_type(8)))  float  v8f;
typedef __attribute__((ext_vector_type(4)))  int    i32x4;

#define BM 128
#define BN 128
#define BK 32

// ---------------------------------------------------------------------------
// Async global->LDS staging (CDNA5 ASYNCcnt path), guarded so the file still
// compiles on toolchains without the builtin.  Builtin signature (from the
// round-2 diagnostic): (int4 AS1*, int4 AS3*, imm offset, imm cpol).
// ---------------------------------------------------------------------------
#if defined(__has_builtin)
#if __has_builtin(__builtin_amdgcn_global_load_async_to_lds_b128) && \
    __has_builtin(__builtin_amdgcn_s_wait_asynccnt)
#define USE_ASYNC_LDS 1
#endif
#endif
#ifndef USE_ASYNC_LDS
#define USE_ASYNC_LDS 0
#endif

#if USE_ASYNC_LDS
typedef i32x4 as1_i32x4 __attribute__((address_space(1)));
typedef i32x4 as3_i32x4 __attribute__((address_space(3)));
#endif

__device__ __forceinline__ void cp_b128(const __bf16* g, __bf16* l) {
#if USE_ASYNC_LDS
    __builtin_amdgcn_global_load_async_to_lds_b128(
        (as1_i32x4*)g, (as3_i32x4*)l, /*offset=*/0, /*cpol=*/0);
#else
    *(bf16x8*)l = *(const bf16x8*)g;
#endif
}

__device__ __forceinline__ void async_wait() {
#if USE_ASYNC_LDS
    __builtin_amdgcn_s_wait_asynccnt(0);
#endif
}

// ---------------------------------------------------------------------------
// WMMA helper: D = A(16x32 bf16) * B(32x16 bf16) + C (f32)
// ---------------------------------------------------------------------------
__device__ __forceinline__ v8f wmma_bf16(v16bf a, v16bf b, v8f c) {
    return __builtin_amdgcn_wmma_f32_16x16x32_bf16(
        false, a, false, b, (short)0, c, false, false);
}

// 16x32 bf16 fragment load (A layout; identical for B when tile is [n][k]):
// lane = row mod 16, half = lane>>4; two contiguous 8-element chunks.
__device__ __forceinline__ v16bf load_frag(const __bf16* base, int row,
                                           int stride, int koff) {
    int lane = threadIdx.x & 31;
    const __bf16* p = base + (size_t)(row + (lane & 15)) * stride + koff + (lane >> 4) * 8;
    union { v16bf v; bf16x8 h[2]; } u;
    u.h[0] = *(const bf16x8*)(p);
    u.h[1] = *(const bf16x8*)(p + 16);
    return u.v;
}

// ---------------------------------------------------------------------------
// Weight prep: fp32 [K][N] -> bf16 transposed [N][K]  (32x32 LDS tiles)
// ---------------------------------------------------------------------------
__global__ __launch_bounds__(256) void transpose_f32_to_bf16_k(
    const float* __restrict__ in, __bf16* __restrict__ out, int K, int N) {
    __shared__ float tile[32][33];
    int n0 = blockIdx.x * 32, k0 = blockIdx.y * 32;
    int tx = threadIdx.x & 31, ty = threadIdx.x >> 5;  // 32 x 8
#pragma unroll
    for (int i = 0; i < 4; i++)
        tile[ty + i * 8][tx] = in[(size_t)(k0 + ty + i * 8) * N + n0 + tx];
    __syncthreads();
#pragma unroll
    for (int i = 0; i < 4; i++)
        out[(size_t)(n0 + ty + i * 8) * K + k0 + tx] = (__bf16)tile[tx][ty + i * 8];
}

// ---------------------------------------------------------------------------
// LayerNorm (D = 1024 fixed), fp32 in -> bf16 out.  One block per row.
// ---------------------------------------------------------------------------
__global__ __launch_bounds__(256) void layernorm_bf16_k(
    const float* __restrict__ x, const float* __restrict__ g,
    const float* __restrict__ b, __bf16* __restrict__ out) {
    int row = blockIdx.x;
    int tid = threadIdx.x;
    int lane = tid & 31, wid = tid >> 5;
    __shared__ float red[8];
    __shared__ float stats[2];

    float4 v = ((const float4*)(x + (size_t)row * 1024))[tid];
    float s = v.x + v.y + v.z + v.w;
#pragma unroll
    for (int m = 16; m >= 1; m >>= 1) s += __shfl_xor(s, m, 32);
    if (lane == 0) red[wid] = s;
    __syncthreads();
    if (tid == 0) {
        float t = 0.f;
#pragma unroll
        for (int i = 0; i < 8; i++) t += red[i];
        stats[0] = t * (1.0f / 1024.0f);
    }
    __syncthreads();
    float mu = stats[0];
    float d0 = v.x - mu, d1 = v.y - mu, d2 = v.z - mu, d3 = v.w - mu;
    float s2 = d0 * d0 + d1 * d1 + d2 * d2 + d3 * d3;
#pragma unroll
    for (int m = 16; m >= 1; m >>= 1) s2 += __shfl_xor(s2, m, 32);
    if (lane == 0) red[wid] = s2;
    __syncthreads();
    if (tid == 0) {
        float t = 0.f;
#pragma unroll
        for (int i = 0; i < 8; i++) t += red[i];
        stats[1] = t * (1.0f / 1024.0f);
    }
    __syncthreads();
    float rinv = rsqrtf(stats[1] + 1e-5f);
    float4 gg = ((const float4*)g)[tid];
    float4 bb = ((const float4*)b)[tid];
    bf16x4 o;
    o[0] = (__bf16)(d0 * rinv * gg.x + bb.x);
    o[1] = (__bf16)(d1 * rinv * gg.y + bb.y);
    o[2] = (__bf16)(d2 * rinv * gg.z + bb.z);
    o[3] = (__bf16)(d3 * rinv * gg.w + bb.w);
    *(bf16x4*)(out + (size_t)row * 1024 + tid * 4) = o;
}

// ---------------------------------------------------------------------------
// bf16 WMMA GEMM, double-buffered async-staged LDS pipeline.
//   C[M,N] = A[M,K] (row-major) * BT[N,K] (pre-transposed weights) + bias
// 256 threads = 8 waves; block tile 128x128, K-tile 32.
// Wave grid 4(M) x 2(N): each wave owns 32x64 = 2x4 WMMA tiles.
// mode 0: fp32 out (+optional residual)     -> Cf
// mode 2: GELU(exact) -> bf16 out           -> Cb
// mode 3: QKV scatter -> bf16 [3][B,H,N,hd] -> Cb
// ---------------------------------------------------------------------------
__global__ __launch_bounds__(256) void gemm_bf16_k(
    const __bf16* __restrict__ A, const __bf16* __restrict__ BT,
    const float* __restrict__ bias, const float* __restrict__ residual,
    float* __restrict__ Cf, __bf16* __restrict__ Cb,
    int M, int N, int K, int mode) {
    __shared__ __attribute__((aligned(16))) __bf16 sA[2][BM * BK];  // [m][k]
    __shared__ __attribute__((aligned(16))) __bf16 sB[2][BN * BK];  // [n][k]

    int tid = threadIdx.x;
    int lane = tid & 31, wid = tid >> 5;
    int bm = blockIdx.x * BM;
    int bn = blockIdx.y * BN;
    int wm = (wid & 3) * 32;
    int wn = (wid >> 2) * 64;

    // stage: both tiles are straight row copies (16B per thread-chunk)
    auto stage = [&](int buf, int k0) {
#pragma unroll
        for (int it = 0; it < 2; it++) {
            int i = tid + it * 256;
            int row = i >> 2;
            int col = (i & 3) * 8;
            cp_b128(&A[(size_t)(bm + row) * K + k0 + col], &sA[buf][row * BK + col]);
            cp_b128(&BT[(size_t)(bn + row) * K + k0 + col], &sB[buf][row * BK + col]);
        }
    };

    v8f acc[2][4] = {};
    int nk = K / BK;
    stage(0, 0);
    for (int kt = 0; kt < nk; kt++) {
        int cur = kt & 1;
        async_wait();        // my async writes for 'cur' complete
        __syncthreads();     // everyone's writes visible; prior readers done
        if (kt + 1 < nk) stage(cur ^ 1, (kt + 1) * BK);

        v16bf af[2], bfg[4];
#pragma unroll
        for (int i = 0; i < 2; i++) af[i] = load_frag(sA[cur], wm + i * 16, BK, 0);
#pragma unroll
        for (int j = 0; j < 4; j++) bfg[j] = load_frag(sB[cur], wn + j * 16, BK, 0);
#pragma unroll
        for (int i = 0; i < 2; i++)
#pragma unroll
            for (int j = 0; j < 4; j++)
                acc[i][j] = wmma_bf16(af[i], bfg[j], acc[i][j]);
    }

    // Epilogue.  C layout: m = vgpr + 8*(lane>>4), n = lane&15.
    int colo = lane & 15;
    int rowo = (lane >> 4) * 8;
#pragma unroll
    for (int i = 0; i < 2; i++) {
#pragma unroll
        for (int j = 0; j < 4; j++) {
#pragma unroll
            for (int r = 0; r < 8; r++) {
                int row = bm + wm + i * 16 + rowo + r;
                int col = bn + wn + j * 16 + colo;
                float val = acc[i][j][r] + bias[col];
                size_t idx = (size_t)row * N + col;
                if (mode == 0) {
                    if (residual) val += residual[idx];
                    Cf[idx] = val;
                } else if (mode == 2) {
                    float ge = 0.5f * val * (1.0f + erff(val * 0.70710678118654752f));
                    Cb[idx] = (__bf16)ge;
                } else {  // mode 3: QKV scatter into [3][B=8,H=16,N=1024,hd=64]
                    int t  = col >> 10;
                    int rm = col & 1023;
                    int hh = rm >> 6;
                    int dd = rm & 63;
                    int b2 = row >> 10;
                    int n2 = row & 1023;
                    Cb[(size_t)t * 8388608 +
                       (((size_t)(b2 * 16 + hh)) * 1024 + n2) * 64 + dd] = (__bf16)val;
                }
            }
        }
    }
}

// ---------------------------------------------------------------------------
// Flash-style attention.  Grid: (8 q-tiles, 16 heads, 8 batches), 256 thr.
// q/k/v: bf16 [B,H,N=1024,hd=64].  Block handles 128 q rows; each wave 16.
// ---------------------------------------------------------------------------
__global__ __launch_bounds__(256) void attention_k(
    const __bf16* __restrict__ q, const __bf16* __restrict__ k,
    const __bf16* __restrict__ v, __bf16* __restrict__ out) {
    __shared__ __attribute__((aligned(16))) __bf16 sK[32 * 64];     // [key][hd]
    __shared__ __attribute__((aligned(16))) __bf16 sVT[64 * 32];    // [hd][key]
    __shared__ __attribute__((aligned(16))) __bf16 sP[8 * 16 * 32]; // per-wave P

    int tid  = threadIdx.x;
    int lane = tid & 31, wid = tid >> 5;
    int head = blockIdx.y, batch = blockIdx.z;
    size_t base = ((size_t)(batch * 16 + head)) * 1024 * 64;
    const __bf16* qp = q + base;
    const __bf16* kp = k + base;
    const __bf16* vp = v + base;

    int qrow = blockIdx.x * 128 + wid * 16;
    v16bf qfrag0 = load_frag(qp, qrow, 64, 0);
    v16bf qfrag1 = load_frag(qp, qrow, 64, 32);

    __bf16* sPw = sP + wid * 512;
    int colo = lane & 15, halfo = lane >> 4;

    float m_i[8], l_i[8];
#pragma unroll
    for (int r = 0; r < 8; r++) { m_i[r] = -3.0e38f; l_i[r] = 0.0f; }
    v8f o[4] = {};

    for (int kt = 0; kt < 32; kt++) {
        __syncthreads();  // prior compute done before overwriting tiles
        {
            int row = tid >> 3;
            int col = (tid & 7) * 8;
            cp_b128(&kp[(size_t)(kt * 32 + row) * 64 + col], &sK[row * 64 + col]);
            bf16x8 t = *(const bf16x8*)&vp[(size_t)(kt * 32 + row) * 64 + col];
#pragma unroll
            for (int j = 0; j < 8; j++) sVT[(col + j) * 32 + row] = t[j];
        }
        async_wait();
        __syncthreads();

        // S = Q * K^T  (16 rows x 32 keys): two 16-key column tiles
        v8f s0 = {}, s1 = {};
        {
            v16bf b0 = load_frag(sK, 0, 64, 0);
            v16bf b1 = load_frag(sK, 0, 64, 32);
            s0 = wmma_bf16(qfrag0, b0, s0);
            s0 = wmma_bf16(qfrag1, b1, s0);
            v16bf b2 = load_frag(sK, 16, 64, 0);
            v16bf b3 = load_frag(sK, 16, 64, 32);
            s1 = wmma_bf16(qfrag0, b2, s1);
            s1 = wmma_bf16(qfrag1, b3, s1);
        }

        // Online softmax per row (row = r + 8*halfo; cols across 16 lanes)
#pragma unroll
        for (int r = 0; r < 8; r++) {
            float a0 = s0[r] * 0.125f;
            float a1 = s1[r] * 0.125f;
            float mx = fmaxf(a0, a1);
#pragma unroll
            for (int m = 8; m >= 1; m >>= 1) mx = fmaxf(mx, __shfl_xor(mx, m, 16));
            float mn   = fmaxf(m_i[r], mx);
            float corr = __expf(m_i[r] - mn);
            float p0   = __expf(a0 - mn);
            float p1   = __expf(a1 - mn);
            float rs   = p0 + p1;
#pragma unroll
            for (int m = 8; m >= 1; m >>= 1) rs += __shfl_xor(rs, m, 16);
            l_i[r] = l_i[r] * corr + rs;
            m_i[r] = mn;
#pragma unroll
            for (int j = 0; j < 4; j++) o[j][r] *= corr;
            int prow = r + 8 * halfo;
            sPw[prow * 32 + colo]      = (__bf16)p0;
            sPw[prow * 32 + 16 + colo] = (__bf16)p1;
        }

        // O += P(16x32) * V(32x64)  — same-wave LDS RAW is in-order (DScnt)
        v16bf pf = load_frag(sPw, 0, 32, 0);
#pragma unroll
        for (int j = 0; j < 4; j++) {
            v16bf vb = load_frag(sVT, j * 16, 32, 0);
            o[j] = wmma_bf16(pf, vb, o[j]);
        }
    }

    // Normalize and write out[B,N,D] as bf16
    size_t outbase = (size_t)batch * 1024 * 1024 + (size_t)head * 64;
#pragma unroll
    for (int j = 0; j < 4; j++) {
#pragma unroll
        for (int r = 0; r < 8; r++) {
            int rr = qrow + r + 8 * halfo;
            float val = o[j][r] / l_i[r];
            out[outbase + (size_t)rr * 1024 + j * 16 + colo] = (__bf16)val;
        }
    }
}

// ---------------------------------------------------------------------------
// Launch sequence
// ---------------------------------------------------------------------------
extern "C" void kernel_launch(void* const* d_in, const int* in_sizes, int n_in,
                              void* d_out, int out_size, void* d_ws, size_t ws_size,
                              hipStream_t stream) {
    const float* x      = (const float*)d_in[0];
    const float* ln1_g  = (const float*)d_in[1];
    const float* ln1_b  = (const float*)d_in[2];
    const float* w_qkv  = (const float*)d_in[3];
    const float* b_qkv  = (const float*)d_in[4];
    const float* w_proj = (const float*)d_in[5];
    const float* b_proj = (const float*)d_in[6];
    const float* ln2_g  = (const float*)d_in[7];
    const float* ln2_b  = (const float*)d_in[8];
    const float* w_fc1  = (const float*)d_in[9];
    const float* b_fc1  = (const float*)d_in[10];
    const float* w_fc2  = (const float*)d_in[11];
    const float* b_fc2  = (const float*)d_in[12];
    float* out = (float*)d_out;

    // Workspace layout (bf16 unless noted); weights stored TRANSPOSED [N][K]
    __bf16* wqkvT  = (__bf16*)d_ws;                        // 3072*1024
    __bf16* wprojT = wqkvT  + (size_t)3072 * 1024;         // 1024*1024
    __bf16* wfc1T  = wprojT + (size_t)1024 * 1024;         // 4096*1024
    __bf16* wfc2T  = wfc1T  + (size_t)4096 * 1024;         // 1024*4096
    __bf16* h1     = wfc2T  + (size_t)1024 * 4096;         // 8192*1024
    __bf16* qb     = h1     + (size_t)8192 * 1024;         // 3 * 8*16*1024*64
    __bf16* attn   = qb     + (size_t)3 * 8388608;         // 8192*1024
    float*  x1     = (float*)(attn + (size_t)8192 * 1024); // 8192*1024 f32
    __bf16* h2     = (__bf16*)(x1 + (size_t)8192 * 1024);  // 8192*1024
    __bf16* m1     = h2     + (size_t)8192 * 1024;         // 8192*4096

    // 1. weights -> bf16 transposed [N][K]
    transpose_f32_to_bf16_k<<<dim3(96, 32),  256, 0, stream>>>(w_qkv,  wqkvT,  1024, 3072);
    transpose_f32_to_bf16_k<<<dim3(32, 32),  256, 0, stream>>>(w_proj, wprojT, 1024, 1024);
    transpose_f32_to_bf16_k<<<dim3(128, 32), 256, 0, stream>>>(w_fc1,  wfc1T,  1024, 4096);
    transpose_f32_to_bf16_k<<<dim3(32, 128), 256, 0, stream>>>(w_fc2,  wfc2T,  4096, 1024);

    // 2. LN1 -> h1 (bf16)
    layernorm_bf16_k<<<8192, 256, 0, stream>>>(x, ln1_g, ln1_b, h1);

    // 3. QKV GEMM with scatter epilogue -> q/k/v [B,H,N,hd]
    gemm_bf16_k<<<dim3(64, 24), 256, 0, stream>>>(h1, wqkvT, b_qkv, nullptr,
                                                  nullptr, qb, 8192, 3072, 1024, 3);

    // 4. Attention -> attn (bf16 [B,N,D])
    attention_k<<<dim3(8, 16, 8), 256, 0, stream>>>(qb, qb + 8388608,
                                                    qb + 2 * 8388608, attn);

    // 5. Proj GEMM + residual(x) -> x1 (fp32)
    gemm_bf16_k<<<dim3(64, 8), 256, 0, stream>>>(attn, wprojT, b_proj, x,
                                                 x1, nullptr, 8192, 1024, 1024, 0);

    // 6. LN2 -> h2 (bf16)
    layernorm_bf16_k<<<8192, 256, 0, stream>>>(x1, ln2_g, ln2_b, h2);

    // 7. FC1 GEMM + exact GELU -> m1 (bf16)
    gemm_bf16_k<<<dim3(64, 32), 256, 0, stream>>>(h2, wfc1T, b_fc1, nullptr,
                                                  nullptr, m1, 8192, 4096, 1024, 2);

    // 8. FC2 GEMM + residual(x1) -> out (fp32)
    gemm_bf16_k<<<dim3(64, 8), 256, 0, stream>>>(m1, wfc2T, b_fc2, x1,
                                                 out, nullptr, 8192, 1024, 4096, 0);
}
